// RULMamba_81750407512800
// MI455X (gfx1250) — compile-verified, matrix operand order checked by reading, not run on hardware
//
#include <hip/hip_runtime.h>
#include <hip/hip_bf16.h>
#include <math.h>

// ---------------------------------------------------------------------------
// RULMamba forward for MI455X (gfx1250, wave32, WMMA).
// All dense matmuls: v_wmma_f32_16x16x32_bf16, bf16 operands pre-converted,
// B pre-transposed to column-major so every fragment load is one b128.
// Each wave computes a 16x64 output strip (4 accumulators per A fragment).
// Decoder layer 0 is dead code in the reference -> only layer 1 computed.
// ---------------------------------------------------------------------------

#define B_SZ 32
#define L_SZ 512
#define M_SZ 16
#define P_SZ 64
#define DMODEL 256
#define DINNER 512
#define DSTATE 16
#define NROWS_ENC (B_SZ * L_SZ)      // 16384
#define LDEC (P_SZ + 1)              // 65
#define NROWS_DEC (B_SZ * LDEC)      // 2080
#define DBCW 64                      // padded dbc width (48 -> 64)

typedef __bf16 bf16_t;
typedef __attribute__((ext_vector_type(8)))  bf16_t v8bf;
typedef __attribute__((ext_vector_type(16))) bf16_t v16bf;
typedef __attribute__((ext_vector_type(8)))  float  v8f;

__device__ __forceinline__ float silu_f(float x) {
    return x * (1.0f / (1.0f + __expf(-x)));
}
__device__ __forceinline__ float sigmoid_f(float x) {
    return 1.0f / (1.0f + __expf(-x));
}

// ---------------------------------------------------------------------------
// GEMM: C[M,N] = act(A[M,K] @ B[K,N] + bias) [* scale]
// A: bf16 row-major (lda). Bt: bf16 column-major (Kp per column), zero-padded.
// Requires M%16==0, N%64==0, K%32==0 (guaranteed by prep padding).
// One wave -> 16x64 strip. act: 0 none, 1 softplus, 2 elu.
// ---------------------------------------------------------------------------
__global__ void wmma_gemm(const bf16_t* __restrict__ A, int lda,
                          const bf16_t* __restrict__ Bt, int kp,
                          const float* __restrict__ bias,
                          float* __restrict__ Cf, int ldc,
                          bf16_t* __restrict__ Ch, int ldch,
                          const float* __restrict__ scale, int rowdiv,
                          int M, int N, int K, int act) {
    const int lane  = threadIdx.x & 31;
    const int wave  = threadIdx.x >> 5;
    const int nsN   = N >> 6;
    const int strip = blockIdx.x * 4 + wave;
    if (strip >= (M >> 4) * nsN) return;      // wave-uniform exit
    const int tm  = strip / nsN;
    const int ts  = strip % nsN;
    const int l15 = lane & 15;
    const int khalf = (lane & 16) ? 8 : 0;

    const bf16_t* Ar = A + (size_t)(tm * 16 + l15) * lda;
    const bf16_t* Bc = Bt + (size_t)(ts * 64 + l15) * kp;

    v8f acc[4] = {v8f{}, v8f{}, v8f{}, v8f{}};
    for (int k0 = 0; k0 < K; k0 += 32) {
        const int kb = k0 + khalf;
        __builtin_prefetch(Ar + kb + 32, 0, 0);
        v8bf a0 = *(const v8bf*)(Ar + kb);
        v8bf a1 = *(const v8bf*)(Ar + kb + 16);
        v16bf af = __builtin_shufflevector(a0, a1,
                       0,1,2,3,4,5,6,7,8,9,10,11,12,13,14,15);
#pragma unroll
        for (int j = 0; j < 4; ++j) {
            const bf16_t* Bj = Bc + (size_t)(j * 16) * kp;
            v8bf b0 = *(const v8bf*)(Bj + kb);
            v8bf b1 = *(const v8bf*)(Bj + kb + 16);
            v16bf bfv = __builtin_shufflevector(b0, b1,
                            0,1,2,3,4,5,6,7,8,9,10,11,12,13,14,15);
            acc[j] = __builtin_amdgcn_wmma_f32_16x16x32_bf16(
                         false, af, false, bfv, (short)0, acc[j], false, false);
        }
    }

    const int rbase = tm * 16 + ((lane & 16) ? 8 : 0);
#pragma unroll
    for (int e = 0; e < 8; ++e) {
        const int r = rbase + e;
#pragma unroll
        for (int j = 0; j < 4; ++j) {
            const int col = ts * 64 + j * 16 + l15;
            float v = acc[j][e] + (bias ? bias[col] : 0.0f);
            if (act == 1)      v = (v > 20.0f) ? v : log1pf(__expf(v));   // softplus
            else if (act == 2) v = (v > 0.0f) ? v : (__expf(v) - 1.0f);   // elu
            if (scale)         v *= scale[(r / rowdiv) * DMODEL + col];
            if (Cf) Cf[(size_t)r * ldc + col] = v;
            if (Ch) Ch[(size_t)r * ldch + col] = (bf16_t)v;
        }
    }
}

// ---------------------------------------------------------------------------
// Prep: pad+convert A (f32 row-major -> bf16 row-major, zero-padded cols)
// ---------------------------------------------------------------------------
__global__ void prep_a(const float* __restrict__ src, int R, int Cs,
                       bf16_t* __restrict__ dst, int Cd) {
    int i = blockIdx.x * blockDim.x + threadIdx.x;
    if (i >= R * Cd) return;
    int c = i % Cd, r = i / Cd;
    dst[i] = (c < Cs) ? (bf16_t)src[(size_t)r * Cs + c] : (bf16_t)0.0f;
}

// Prep: transpose+convert B (f32 [K x N] row-major -> bf16 col-major, Kp rows)
__global__ void prep_bt(const float* __restrict__ src, int K, int N, int ldb,
                        bf16_t* __restrict__ dst, int Kp, int Np) {
    int i = blockIdx.x * blockDim.x + threadIdx.x;
    if (i >= Kp * Np) return;
    int k = i % Kp, c = i / Kp;
    dst[i] = (k < K && c < N) ? (bf16_t)src[(size_t)k * ldb + c] : (bf16_t)0.0f;
}

// ---------------------------------------------------------------------------
// SE path (tiny): scale = sigmoid(relu(pooled@W1)@W2); ve_b column sums.
// ---------------------------------------------------------------------------
__global__ void se_kernel(const float* __restrict__ x_enc,
                          const float* __restrict__ ve_W,
                          const float* __restrict__ ve_b,
                          const float* __restrict__ W1,
                          const float* __restrict__ W2,
                          float* __restrict__ scale,
                          float* __restrict__ bias_sum) {
    __shared__ float sx[M_SZ];
    __shared__ float pooled[DMODEL];
    __shared__ float hid[DMODEL / 2];
    const int b = blockIdx.x, t = threadIdx.x;
    if (t < M_SZ) {
        float s = 0.0f;
        for (int l = 0; l < L_SZ; ++l)
            s += x_enc[((size_t)b * L_SZ + l) * M_SZ + t];
        sx[t] = s * (1.0f / L_SZ);
    }
    __syncthreads();
    {
        float p = 0.0f;
        for (int m = 0; m < M_SZ; ++m)
            p += sx[m] * ve_W[m * DMODEL + t] + ve_b[m * DMODEL + t];
        pooled[t] = p * (1.0f / M_SZ);
    }
    __syncthreads();
    if (t < DMODEL / 2) {
        float s = 0.0f;
        for (int c = 0; c < DMODEL; ++c) s += pooled[c] * W1[c * (DMODEL / 2) + t];
        hid[t] = s > 0.0f ? s : 0.0f;
    }
    __syncthreads();
    {
        float s = 0.0f;
        for (int j = 0; j < DMODEL / 2; ++j) s += hid[j] * W2[j * DMODEL + t];
        scale[b * DMODEL + t] = sigmoid_f(s);
    }
    if (b == 0) {
        float s = 0.0f;
        for (int m = 0; m < M_SZ; ++m) s += ve_b[m * DMODEL + t];
        bias_sum[t] = s;
    }
}

// Depthwise causal conv4 + SiLU; writes f32 (for scan) and bf16 (GEMM A)
__global__ void conv_silu(const float* __restrict__ xz,
                          const float* __restrict__ cw,
                          const float* __restrict__ cb,
                          float* __restrict__ outf,
                          bf16_t* __restrict__ outh, int Bb, int Ls) {
    int i = blockIdx.x * blockDim.x + threadIdx.x;
    if (i >= Bb * Ls * DINNER) return;
    int d = i % DINNER;
    int t = (i / DINNER) % Ls;
    int b = i / (DINNER * Ls);
    const float* xi = xz + (size_t)b * Ls * (2 * DINNER);
    float acc = cb[d];
#pragma unroll
    for (int k = 0; k < 4; ++k) {
        int tt = t + k - 3;
        if (tt >= 0) acc += xi[(size_t)tt * (2 * DINNER) + d] * cw[d * 4 + k];
    }
    float v = silu_f(acc);
    outf[i] = v;
    outh[i] = (bf16_t)v;
}

// Selective scan; fused +x*D and *silu(res); writes bf16 for the out-proj GEMM
__global__ void scan_kernel(const float* __restrict__ delta,
                            const float* __restrict__ xi,
                            const float* __restrict__ dbc,
                            const float* __restrict__ xz,
                            const float* __restrict__ A_log,
                            const float* __restrict__ Dp,
                            bf16_t* __restrict__ yout, int Bb, int Ls) {
    int i = blockIdx.x * blockDim.x + threadIdx.x;
    if (i >= Bb * DINNER) return;
    int d = i % DINNER, b = i / DINNER;
    float Arow[DSTATE];
#pragma unroll
    for (int n = 0; n < DSTATE; ++n) Arow[n] = -__expf(A_log[d * DSTATE + n]);
    const float Dd = Dp[d];
    float h[DSTATE];
#pragma unroll
    for (int n = 0; n < DSTATE; ++n) h[n] = 0.0f;
    const size_t base = (size_t)b * Ls;
    for (int t = 0; t < Ls; ++t) {
        const size_t r = base + t;
        float dt = delta[r * DINNER + d];
        float x  = xi[r * DINNER + d];
        const float* bc = dbc + r * DBCW;
        float y = 0.0f;
#pragma unroll
        for (int n = 0; n < DSTATE; ++n) {
            float dA = __expf(dt * Arow[n]);
            h[n] = dA * h[n] + dt * bc[16 + n] * x;
            y += h[n] * bc[32 + n];
        }
        float res = xz[r * (2 * DINNER) + DINNER + d];
        yout[r * DINNER + d] = (bf16_t)((y + x * Dd) * silu_f(res));
    }
}

// xin = concat([x_dec, context], axis=1) -> bf16 (GEMM A / concat source)
__global__ void build_xin(const float* __restrict__ x_dec,
                          const float* __restrict__ enc_out,
                          bf16_t* __restrict__ xin) {
    int i = blockIdx.x * blockDim.x + threadIdx.x;
    if (i >= NROWS_DEC * DMODEL) return;
    int c = i % DMODEL;
    int r = i / DMODEL;
    int b = r / LDEC, p = r % LDEC;
    float v = (p < P_SZ) ? x_dec[((size_t)b * P_SZ + p) * DMODEL + c]
                         : enc_out[((size_t)b * L_SZ + (L_SZ - 1)) * DMODEL + c];
    xin[i] = (bf16_t)v;
}

// cat = [mamba_out, xin] along features (bf16 -> bf16)
__global__ void build_cat(const bf16_t* __restrict__ y,
                          const bf16_t* __restrict__ xin,
                          bf16_t* __restrict__ cat) {
    int i = blockIdx.x * blockDim.x + threadIdx.x;
    if (i >= NROWS_DEC * 2 * DMODEL) return;
    int c = i % (2 * DMODEL);
    int r = i / (2 * DMODEL);
    cat[i] = (c < DMODEL) ? y[(size_t)r * DMODEL + c]
                          : xin[(size_t)r * DMODEL + (c - DMODEL)];
}

// out = LayerNorm(s + a*sigmoid(bg)) * g + b ; block per row, wave32 reduce
__global__ void grn_ln(const float* __restrict__ s, const float* __restrict__ a,
                       const float* __restrict__ bg,
                       const float* __restrict__ ln_g, const float* __restrict__ ln_b,
                       float* __restrict__ out) {
    __shared__ float red1[8];
    __shared__ float red2[8];
    const int r = blockIdx.x, c = threadIdx.x;
    const size_t idx = (size_t)r * DMODEL + c;
    float v = s[idx] + a[idx] * sigmoid_f(bg[idx]);
    float p = v;
    for (int off = 16; off > 0; off >>= 1) p += __shfl_down(p, off);
    if ((c & 31) == 0) red1[c >> 5] = p;
    __syncthreads();
    if (c == 0) {
        float m = 0.0f;
        for (int i2 = 0; i2 < 8; ++i2) m += red1[i2];
        red1[0] = m * (1.0f / DMODEL);
    }
    __syncthreads();
    const float mean = red1[0];
    float d = v - mean;
    float q = d * d;
    for (int off = 16; off > 0; off >>= 1) q += __shfl_down(q, off);
    if ((c & 31) == 0) red2[c >> 5] = q;
    __syncthreads();
    if (c == 0) {
        float m = 0.0f;
        for (int i2 = 0; i2 < 8; ++i2) m += red2[i2];
        red2[0] = m * (1.0f / DMODEL);
    }
    __syncthreads();
    out[idx] = d * rsqrtf(red2[0] + 1e-5f) * ln_g[c] + ln_b[c];
}

// Final head
__global__ void final_proj(const float* __restrict__ grn_out,
                           const float* __restrict__ pW,
                           const float* __restrict__ pb,
                           float* __restrict__ out) {
    int i = blockIdx.x * blockDim.x + threadIdx.x;
    if (i >= B_SZ * P_SZ) return;
    int p = i % P_SZ, b = i / P_SZ;
    const float* row = grn_out + ((size_t)b * LDEC + 1 + p) * DMODEL;
    float acc = pb[0];
    for (int c = 0; c < DMODEL; ++c) acc += row[c] * pW[c];
    out[i] = acc;
}

// ---------------------------------------------------------------------------
// Host launcher
// ---------------------------------------------------------------------------
struct MambaP {
    const float *in_W, *in_b, *conv_W, *conv_b, *xproj_W, *dt_W, *dt_b,
                *A_log, *D, *out_W, *out_b;
};
static MambaP get_mamba(void* const* d_in, int base) {
    MambaP m;
    m.in_W    = (const float*)d_in[base + 0];
    m.in_b    = (const float*)d_in[base + 1];
    m.conv_W  = (const float*)d_in[base + 2];
    m.conv_b  = (const float*)d_in[base + 3];
    m.xproj_W = (const float*)d_in[base + 4];
    m.dt_W    = (const float*)d_in[base + 5];
    m.dt_b    = (const float*)d_in[base + 6];
    m.A_log   = (const float*)d_in[base + 7];
    m.D       = (const float*)d_in[base + 8];
    m.out_W   = (const float*)d_in[base + 9];
    m.out_b   = (const float*)d_in[base + 10];
    return m;
}

static inline void gemm(hipStream_t st, const bf16_t* A, int lda,
                        const bf16_t* Bt, int kp, const float* bias,
                        float* Cf, int ldc, bf16_t* Ch, int ldch,
                        const float* scale, int rowdiv,
                        int M, int N, int K, int act) {
    int strips = (M / 16) * (N / 64);
    wmma_gemm<<<(strips + 3) / 4, 128, 0, st>>>(A, lda, Bt, kp, bias, Cf, ldc,
                                                Ch, ldch, scale, rowdiv, M, N, K, act);
}
static inline void bt(hipStream_t st, const float* src, int K, int N, int ldb,
                      bf16_t* dst, int Kp, int Np) {
    int n = Kp * Np;
    prep_bt<<<(n + 255) / 256, 256, 0, st>>>(src, K, N, ldb, dst, Kp, Np);
}

extern "C" void kernel_launch(void* const* d_in, const int* in_sizes, int n_in,
                              void* d_out, int out_size, void* d_ws, size_t ws_size,
                              hipStream_t stream) {
    const float* x_enc = (const float*)d_in[0];
    const float* x_dec = (const float*)d_in[2];
    const float* ve_W  = (const float*)d_in[4];
    const float* ve_b  = (const float*)d_in[5];
    const float* se_W1 = (const float*)d_in[6];
    const float* se_W2 = (const float*)d_in[7];
    MambaP enc = get_mamba(d_in, 8);
    MambaP dec = get_mamba(d_in, 40);     // layer 0 (19..39) is dead code
    const float* gW1  = (const float*)d_in[51];
    const float* gb1  = (const float*)d_in[52];
    const float* gW2  = (const float*)d_in[53];
    const float* gb2  = (const float*)d_in[54];
    const float* g1W  = (const float*)d_in[55];
    const float* g1b  = (const float*)d_in[56];
    const float* g2W  = (const float*)d_in[57];
    const float* g2b  = (const float*)d_in[58];
    const float* ln_g = (const float*)d_in[59];
    const float* ln_b = (const float*)d_in[60];
    const float* pW   = (const float*)d_in[61];
    const float* pb   = (const float*)d_in[62];
    float* out = (float*)d_out;

    // ---- workspace (byte cursor, 256B aligned blocks) ----
    char* base = (char*)d_ws;
    size_t cur = 0;
    auto alloc = [&](size_t bytes) -> char* {
        char* p = base + cur;
        cur += (bytes + 255) & ~(size_t)255;
        return p;
    };
    // persistent
    float*  scale    = (float*)alloc(B_SZ * DMODEL * 4);
    float*  bias_sum = (float*)alloc(DMODEL * 4);
    bf16_t* xenc_pad = (bf16_t*)alloc((size_t)NROWS_ENC * 32 * 2);
    bf16_t* veW_t    = (bf16_t*)alloc(32 * DMODEL * 2);
    bf16_t* inWt_e   = (bf16_t*)alloc((size_t)DMODEL * 2 * DINNER * 2);
    bf16_t* xprojWt_e= (bf16_t*)alloc((size_t)DINNER * DBCW * 2);
    bf16_t* dtWt_e   = (bf16_t*)alloc(32 * DINNER * 2);
    bf16_t* outWt_e  = (bf16_t*)alloc((size_t)DINNER * DMODEL * 2);
    bf16_t* inWt_d   = (bf16_t*)alloc((size_t)DMODEL * 2 * DINNER * 2);
    bf16_t* xprojWt_d= (bf16_t*)alloc((size_t)DINNER * DBCW * 2);
    bf16_t* dtWt_d   = (bf16_t*)alloc(32 * DINNER * 2);
    bf16_t* outWt_d  = (bf16_t*)alloc((size_t)DINNER * DMODEL * 2);
    bf16_t* W1t      = (bf16_t*)alloc((size_t)2 * DMODEL * 2 * DMODEL * 2);
    bf16_t* W2t      = (bf16_t*)alloc((size_t)2 * DMODEL * DMODEL * 2);
    bf16_t* g1Wt     = (bf16_t*)alloc((size_t)DMODEL * DMODEL * 2);
    bf16_t* g2Wt     = (bf16_t*)alloc((size_t)DMODEL * DMODEL * 2);
    float*  enc_out  = (float*)alloc((size_t)NROWS_ENC * DMODEL * 4);
    bf16_t* xin_bf   = (bf16_t*)alloc((size_t)NROWS_DEC * DMODEL * 2);
    // phase arena
    size_t arena = cur;
    bf16_t* encin_bf = (bf16_t*)alloc((size_t)NROWS_ENC * DMODEL * 2);
    float*  xz_e     = (float*)alloc((size_t)NROWS_ENC * 2 * DINNER * 4);
    float*  xi_ef    = (float*)alloc((size_t)NROWS_ENC * DINNER * 4);
    bf16_t* xi_eh    = (bf16_t*)alloc((size_t)NROWS_ENC * DINNER * 2);
    float*  dbc_ef   = (float*)alloc((size_t)NROWS_ENC * DBCW * 4);
    bf16_t* dbc_eh   = (bf16_t*)alloc((size_t)NROWS_ENC * DBCW * 2);
    float*  delta_e  = (float*)alloc((size_t)NROWS_ENC * DINNER * 4);
    bf16_t* y_eh     = (bf16_t*)alloc((size_t)NROWS_ENC * DINNER * 2);
    // decoder phase reuses the arena
    cur = arena;
    float*  xz_d     = (float*)alloc((size_t)NROWS_DEC * 2 * DINNER * 4);
    float*  xi_df    = (float*)alloc((size_t)NROWS_DEC * DINNER * 4);
    bf16_t* xi_dh    = (bf16_t*)alloc((size_t)NROWS_DEC * DINNER * 2);
    float*  dbc_df   = (float*)alloc((size_t)NROWS_DEC * DBCW * 4);
    bf16_t* dbc_dh   = (bf16_t*)alloc((size_t)NROWS_DEC * DBCW * 2);
    float*  delta_d  = (float*)alloc((size_t)NROWS_DEC * DINNER * 4);
    bf16_t* y_dh     = (bf16_t*)alloc((size_t)NROWS_DEC * DINNER * 2);
    float*  mo_f     = (float*)alloc((size_t)NROWS_DEC * DMODEL * 4);
    bf16_t* mo_h     = (bf16_t*)alloc((size_t)NROWS_DEC * DMODEL * 2);
    bf16_t* cat_h    = (bf16_t*)alloc((size_t)NROWS_DEC * 2 * DMODEL * 2);
    bf16_t* h1_h     = (bf16_t*)alloc((size_t)NROWS_DEC * 2 * DMODEL * 2);
    bf16_t* h_h      = (bf16_t*)alloc((size_t)NROWS_DEC * DMODEL * 2);
    float*  a_f      = (float*)alloc((size_t)NROWS_DEC * DMODEL * 4);
    float*  bg_f     = (float*)alloc((size_t)NROWS_DEC * DMODEL * 4);
    float*  grn_f    = (float*)alloc((size_t)NROWS_DEC * DMODEL * 4);

    // ---- prep: pad/convert/transpose (once per launch, tiny) ----
    {
        int n = NROWS_ENC * 32;
        prep_a<<<(n + 255) / 256, 256, 0, stream>>>(x_enc, NROWS_ENC, M_SZ, xenc_pad, 32);
    }
    bt(stream, ve_W,        M_SZ,  DMODEL,     DMODEL,     veW_t,     32,  DMODEL);
    bt(stream, enc.in_W,    DMODEL, 2*DINNER,  2*DINNER,   inWt_e,    DMODEL, 2*DINNER);
    bt(stream, enc.xproj_W, DINNER, 48,        48,         xprojWt_e, DINNER, DBCW);
    bt(stream, enc.dt_W,    M_SZ,  DINNER,     DINNER,     dtWt_e,    32,  DINNER);
    bt(stream, enc.out_W,   DINNER, DMODEL,    DMODEL,     outWt_e,   DINNER, DMODEL);
    bt(stream, dec.in_W,    DMODEL, 2*DINNER,  2*DINNER,   inWt_d,    DMODEL, 2*DINNER);
    bt(stream, dec.xproj_W, DINNER, 48,        48,         xprojWt_d, DINNER, DBCW);
    bt(stream, dec.dt_W,    M_SZ,  DINNER,     DINNER,     dtWt_d,    32,  DINNER);
    bt(stream, dec.out_W,   DINNER, DMODEL,    DMODEL,     outWt_d,   DINNER, DMODEL);
    bt(stream, gW1,  2*DMODEL, 2*DMODEL, 2*DMODEL, W1t,  2*DMODEL, 2*DMODEL);
    bt(stream, gW2,  2*DMODEL, DMODEL,   DMODEL,   W2t,  2*DMODEL, DMODEL);
    bt(stream, g1W,  DMODEL,   DMODEL,   DMODEL,   g1Wt, DMODEL,   DMODEL);
    bt(stream, g2W,  DMODEL,   DMODEL,   DMODEL,   g2Wt, DMODEL,   DMODEL);

    // ================= encoder =================
    se_kernel<<<B_SZ, DMODEL, 0, stream>>>(x_enc, ve_W, ve_b, se_W1, se_W2, scale, bias_sum);
    // enc_in = (x_enc @ ve_W + sum_m ve_b) * scale  (scale fused in epilogue)
    gemm(stream, xenc_pad, 32, veW_t, 32, bias_sum, nullptr, 0, encin_bf, DMODEL,
         scale, L_SZ, NROWS_ENC, DMODEL, 32, 0);
    gemm(stream, encin_bf, DMODEL, inWt_e, DMODEL, enc.in_b, xz_e, 2*DINNER,
         nullptr, 0, nullptr, 0, NROWS_ENC, 2*DINNER, DMODEL, 0);
    {
        int n = NROWS_ENC * DINNER;
        conv_silu<<<(n + 255) / 256, 256, 0, stream>>>(xz_e, enc.conv_W, enc.conv_b,
                                                       xi_ef, xi_eh, B_SZ, L_SZ);
    }
    gemm(stream, xi_eh, DINNER, xprojWt_e, DINNER, nullptr, dbc_ef, DBCW,
         dbc_eh, DBCW, nullptr, 0, NROWS_ENC, DBCW, DINNER, 0);
    gemm(stream, dbc_eh, DBCW, dtWt_e, 32, enc.dt_b, delta_e, DINNER,
         nullptr, 0, nullptr, 0, NROWS_ENC, DINNER, 32, 1);           // softplus
    {
        int n = B_SZ * DINNER;
        scan_kernel<<<(n + 255) / 256, 256, 0, stream>>>(delta_e, xi_ef, dbc_ef, xz_e,
                                                         enc.A_log, enc.D, y_eh, B_SZ, L_SZ);
    }
    gemm(stream, y_eh, DINNER, outWt_e, DINNER, enc.out_b, enc_out, DMODEL,
         nullptr, 0, nullptr, 0, NROWS_ENC, DMODEL, DINNER, 0);

    // ================= decoder (last layer only) =================
    {
        int n = NROWS_DEC * DMODEL;
        build_xin<<<(n + 255) / 256, 256, 0, stream>>>(x_dec, enc_out, xin_bf);
    }
    gemm(stream, xin_bf, DMODEL, inWt_d, DMODEL, dec.in_b, xz_d, 2*DINNER,
         nullptr, 0, nullptr, 0, NROWS_DEC, 2*DINNER, DMODEL, 0);
    {
        int n = NROWS_DEC * DINNER;
        conv_silu<<<(n + 255) / 256, 256, 0, stream>>>(xz_d, dec.conv_W, dec.conv_b,
                                                       xi_df, xi_dh, B_SZ, LDEC);
    }
    gemm(stream, xi_dh, DINNER, xprojWt_d, DINNER, nullptr, dbc_df, DBCW,
         dbc_dh, DBCW, nullptr, 0, NROWS_DEC, DBCW, DINNER, 0);
    gemm(stream, dbc_dh, DBCW, dtWt_d, 32, dec.dt_b, delta_d, DINNER,
         nullptr, 0, nullptr, 0, NROWS_DEC, DINNER, 32, 1);           // softplus
    {
        int n = B_SZ * DINNER;
        scan_kernel<<<(n + 255) / 256, 256, 0, stream>>>(delta_d, xi_df, dbc_df, xz_d,
                                                         dec.A_log, dec.D, y_dh, B_SZ, LDEC);
    }
    gemm(stream, y_dh, DINNER, outWt_d, DINNER, dec.out_b, mo_f, DMODEL,
         mo_h, DMODEL, nullptr, 0, NROWS_DEC, DMODEL, DINNER, 0);
    {
        int n = NROWS_DEC * 2 * DMODEL;
        build_cat<<<(n + 255) / 256, 256, 0, stream>>>(mo_h, xin_bf, cat_h);
    }
    gemm(stream, cat_h, 2*DMODEL, W1t, 2*DMODEL, gb1, nullptr, 0,
         h1_h, 2*DMODEL, nullptr, 0, NROWS_DEC, 2*DMODEL, 2*DMODEL, 2);  // elu
    gemm(stream, h1_h, 2*DMODEL, W2t, 2*DMODEL, gb2, nullptr, 0,
         h_h, DMODEL, nullptr, 0, NROWS_DEC, DMODEL, 2*DMODEL, 0);
    gemm(stream, h_h, DMODEL, g1Wt, DMODEL, g1b, a_f, DMODEL,
         nullptr, 0, nullptr, 0, NROWS_DEC, DMODEL, DMODEL, 0);
    gemm(stream, h_h, DMODEL, g2Wt, DMODEL, g2b, bg_f, DMODEL,
         nullptr, 0, nullptr, 0, NROWS_DEC, DMODEL, DMODEL, 0);
    grn_ln<<<NROWS_DEC, DMODEL, 0, stream>>>(mo_f, a_f, bg_f, ln_g, ln_b, grn_f);
    {
        int n = B_SZ * P_SZ;
        final_proj<<<(n + 255) / 256, 256, 0, stream>>>(grn_f, pW, pb, out);
    }
}
